// CIN0_PH_54803782697123
// MI455X (gfx1250) — matrix operand order repaired and checked by reading, not compiled, not run on hardware
//
#include <hip/hip_runtime.h>

// ---------------------------------------------------------------------------
// CDNA5 (gfx1250) implementation of the CIN0+PH network.
// Heavy GEMMs use v_wmma_f32_16x16x32_bf16 (wave32, 16x16 tiles, f32 accum).
// Each wave runs TWO independent accumulator chains (two M-tiles) so the
// XDL pipe overlaps WMMA latency, and each B (weight) fragment is reused.
// ---------------------------------------------------------------------------

typedef __attribute__((ext_vector_type(16))) __bf16 bf16x16;
typedef __attribute__((ext_vector_type(8)))  __bf16 bf16x8;
typedef __attribute__((ext_vector_type(8)))  float  f32x8;

static constexpr int cN0 = 100000, cN1 = 200000, cN2 = 50000;
static constexpr int cF  = 128, cB = 256;
static constexpr int cNL = 3, cNF = 8, cFH = 16, cOPH = 64, cNC = 10;
static constexpr int cE0U = 400000, cE1U = 400000, cE1D = 400000;
static constexpr int cE2D = 100000, cEG  = 200000;

// ---------------- WMMA fragment helpers (contiguous k, same order in A and B) --

__device__ __forceinline__ bf16x16 load_a_frag(const __bf16* row, int kk, int half) {
    // A 16x32 bf16: lane holds row M=lane&15; K = half*16 + i  (16 contiguous)
    const __bf16* p = row + kk * 32 + half * 16;
    bf16x8 lo = *reinterpret_cast<const bf16x8*>(p);
    bf16x8 hi = *reinterpret_cast<const bf16x8*>(p + 8);
    bf16x16 a;
#pragma unroll
    for (int i = 0; i < 8; ++i) { a[i] = lo[i]; a[i + 8] = hi[i]; }
    return a;
}

__device__ __forceinline__ bf16x16 load_b_frag(const __bf16* p) {
    // B 32x16 bf16: lane holds col N=lane&15; K = half*16 + i  (16 contiguous)
    bf16x8 lo = *reinterpret_cast<const bf16x8*>(p);
    bf16x8 hi = *reinterpret_cast<const bf16x8*>(p + 8);
    bf16x16 b;
#pragma unroll
    for (int i = 0; i < 8; ++i) { b[i] = lo[i]; b[i + 8] = hi[i]; }
    return b;
}

__device__ __forceinline__ f32x8 wmma_bf16(bf16x16 a, bf16x16 b, f32x8 c) {
    return __builtin_amdgcn_wmma_f32_16x16x32_bf16(false, a, false, b, (short)0, c,
                                                   false, false);
}

// ---------------- Edge-message GEMM: relu(cat(x[src],y[cof]) @ W + b) -------
// 32 edges per block (two M-tiles per wave -> 2 independent WMMA chains).
// Scatters raw message to acc[dst], counts edges per dst, accumulates
// per-feature sum / sumsq for the (affine) BatchNorm folded in later.

__global__ __launch_bounds__(256) void msg_gemm_kernel(
    const float* __restrict__ Xsrc, const float* __restrict__ Xcof,
    const int* __restrict__ src, const int* __restrict__ dst,
    const int* __restrict__ cof, int E,
    const __bf16* __restrict__ Wt, const float* __restrict__ bias,
    float* __restrict__ acc, float* __restrict__ cnt,
    float* __restrict__ ssum, float* __restrict__ ssq)
{
    __shared__ __align__(16) __bf16 As[32][264];   // 32 edges x 256 (padded)
    __shared__ int dsts[32];
    const int t = threadIdx.x;
    const int tile = blockIdx.x;
    {
        const int row = t >> 4, chunk = t & 15;
#pragma unroll
        for (int tt = 0; tt < 2; ++tt) {
            const int rr = row + tt * 16;
            int e = tile * 32 + rr;
            if (e >= E) e = E - 1;
            const float* rp = (chunk < 8)
                ? Xsrc + (size_t)src[e] * cF + chunk * 16
                : Xcof + (size_t)cof[e] * cF + (chunk - 8) * 16;
#pragma unroll
            for (int j = 0; j < 4; ++j) {
                const float4 v = *reinterpret_cast<const float4*>(rp + j * 4);
                const int c0 = chunk * 16 + j * 4;
                As[rr][c0 + 0] = (__bf16)v.x;
                As[rr][c0 + 1] = (__bf16)v.y;
                As[rr][c0 + 2] = (__bf16)v.z;
                As[rr][c0 + 3] = (__bf16)v.w;
            }
            if (chunk == 0) {
                dsts[rr] = dst[e];
                if (tile * 32 + rr < E) atomicAdd(&cnt[dst[e]], 1.0f);
            }
        }
    }
    __syncthreads();

    const int wave = t >> 5, lane = t & 31;
    const int n = wave * 16 + (lane & 15);
    const int half = lane >> 4;
    f32x8 c0 = {}, c1 = {};
#pragma unroll
    for (int kk = 0; kk < 8; ++kk) {               // K = 256
        bf16x16 a0 = load_a_frag(&As[lane & 15][0], kk, half);
        bf16x16 a1 = load_a_frag(&As[16 + (lane & 15)][0], kk, half);
        bf16x16 b  = load_b_frag(Wt + (size_t)n * 256 + kk * 32 + half * 16);
        c0 = wmma_bf16(a0, b, c0);                 // two independent chains
        c1 = wmma_bf16(a1, b, c1);
    }
    const float bcol = bias[n];
    float ps = 0.f, pq = 0.f;
#pragma unroll
    for (int tt = 0; tt < 2; ++tt) {
        const f32x8 cc = tt ? c1 : c0;
#pragma unroll
        for (int j = 0; j < 8; ++j) {
            const int m = tt * 16 + j + 8 * half;
            const int e = tile * 32 + m;
            float v = cc[j] + bcol;
            v = v > 0.f ? v : 0.f;
            if (e < E) {
                ps += v; pq += v * v;
                atomicAdd(&acc[(size_t)dsts[m] * cF + n], v);
            }
        }
    }
    ps += __shfl_xor(ps, 16);
    pq += __shfl_xor(pq, 16);
    if (half == 0) { atomicAdd(&ssum[n], ps); atomicAdd(&ssq[n], pq); }
}

// ---------------- BN affine fit: s = g*rsqrt(var+eps), t = be - mean*s ------
__global__ void bnfit_kernel(float* __restrict__ slot,   // [ssum|ssq|s|t] x128
                             const float* __restrict__ g,
                             const float* __restrict__ be, float inv)
{
    const int f = threadIdx.x;
    const float m = slot[f] * inv;
    const float v = slot[128 + f] * inv - m * m;
    const float s = g[f] * rsqrtf(v + 1e-5f);
    slot[256 + f] = s;
    slot[384 + f] = be[f] - m * s;
}

// ---------------- Fused update MLP: relu(W2·relu(W1·(x + Σ affine(acc)))) ---
// 32 rows per block; both GEMM stages run dual accumulator chains.
__global__ __launch_bounds__(256) void upd_gemm_kernel(
    const float* __restrict__ X, int N,
    const float* __restrict__ accA, const float* __restrict__ cntA,
    const float* __restrict__ stA,                         // s@0, t@128
    const float* __restrict__ accB, const float* __restrict__ cntB,
    const float* __restrict__ stB,                         // nullable
    const __bf16* __restrict__ W1t, const float* __restrict__ b1,
    const __bf16* __restrict__ W2t, const float* __restrict__ b2,
    float* __restrict__ out,                               // pre-BN h2
    float* __restrict__ ssum, float* __restrict__ ssq)
{
    __shared__ __align__(16) __bf16 As[32][136];
    __shared__ __align__(16) __bf16 Hs[32][136];
    const int t = threadIdx.x, tile = blockIdx.x;
    {
        const int row = t >> 4, chunk = t & 15;
#pragma unroll
        for (int tt = 0; tt < 2; ++tt) {
            const int rr = row + tt * 16;
            int i = tile * 32 + rr;
            if (i >= N) i = N - 1;
            const size_t base = (size_t)i * cF + chunk * 8;
            const float ca = cntA[i];
            const float cb = accB ? cntB[i] : 0.f;
#pragma unroll
            for (int j = 0; j < 8; ++j) {
                const int col = chunk * 8 + j;
                float y = X[base + j] + stA[col] * accA[base + j] + stA[128 + col] * ca;
                if (accB) y += stB[col] * accB[base + j] + stB[128 + col] * cb;
                As[rr][col] = (__bf16)y;
            }
        }
    }
    __syncthreads();

    const int wave = t >> 5, lane = t & 31;
    const int n = wave * 16 + (lane & 15);
    const int half = lane >> 4;

    f32x8 c0 = {}, c1 = {};
#pragma unroll
    for (int kk = 0; kk < 4; ++kk) {               // K = 128
        bf16x16 a0 = load_a_frag(&As[lane & 15][0], kk, half);
        bf16x16 a1 = load_a_frag(&As[16 + (lane & 15)][0], kk, half);
        bf16x16 b  = load_b_frag(W1t + (size_t)n * 128 + kk * 32 + half * 16);
        c0 = wmma_bf16(a0, b, c0);
        c1 = wmma_bf16(a1, b, c1);
    }
    const float bc1 = b1[n];
#pragma unroll
    for (int tt = 0; tt < 2; ++tt) {
        const f32x8 cc = tt ? c1 : c0;
#pragma unroll
        for (int j = 0; j < 8; ++j) {
            const int m = tt * 16 + j + 8 * half;
            float v = cc[j] + bc1;
            Hs[m][n] = (__bf16)(v > 0.f ? v : 0.f);
        }
    }
    __syncthreads();

    f32x8 d0 = {}, d1 = {};
#pragma unroll
    for (int kk = 0; kk < 4; ++kk) {
        bf16x16 a0 = load_a_frag(&Hs[lane & 15][0], kk, half);
        bf16x16 a1 = load_a_frag(&Hs[16 + (lane & 15)][0], kk, half);
        bf16x16 b  = load_b_frag(W2t + (size_t)n * 128 + kk * 32 + half * 16);
        d0 = wmma_bf16(a0, b, d0);
        d1 = wmma_bf16(a1, b, d1);
    }
    const float bc2 = b2[n];
    float ps = 0.f, pq = 0.f;
#pragma unroll
    for (int tt = 0; tt < 2; ++tt) {
        const f32x8 cc = tt ? d1 : d0;
#pragma unroll
        for (int j = 0; j < 8; ++j) {
            const int m = tt * 16 + j + 8 * half;
            const int i = tile * 32 + m;
            float v = cc[j] + bc2;
            v = v > 0.f ? v : 0.f;
            if (i < N) {
                out[(size_t)i * cF + n] = v;
                ps += v; pq += v * v;
            }
        }
    }
    ps += __shfl_xor(ps, 16);
    pq += __shfl_xor(pq, 16);
    if (half == 0) { atomicAdd(&ssum[n], ps); atomicAdd(&ssq[n], pq); }
}

__global__ void bn_apply_kernel(const float* __restrict__ h, float* __restrict__ X,
                                const float* __restrict__ st, size_t total)
{
    const size_t idx = (size_t)blockIdx.x * 256 + threadIdx.x;
    if (idx >= total) return;
    const int f = (int)(idx & 127);
    X[idx] = h[idx] * st[f] + st[128 + f];
}

// ---------------- PH: sigmoid filtration v = sig(relu(x@fW1+b)@fW2+b) -------
__global__ __launch_bounds__(256) void filt_kernel(
    const float* __restrict__ X0, int N,
    const float* __restrict__ fW1, const float* __restrict__ fb1,
    const float* __restrict__ fW2, const float* __restrict__ fb2,
    float* __restrict__ V)
{
    __shared__ float W1s[128 * 16];
    __shared__ float W2s[16 * 8];
    __shared__ float b1s[16], b2s[8];
    const int t = threadIdx.x;
    for (int i = t; i < 128 * 16; i += 256) W1s[i] = fW1[i];
    if (t < 16 * 8) W2s[t] = fW2[t];
    if (t < 16) b1s[t] = fb1[t];
    if (t < 8)  b2s[t] = fb2[t];
    __syncthreads();
    const int i = blockIdx.x * 256 + t;
    if (i >= N) return;
    float h[16];
#pragma unroll
    for (int j = 0; j < 16; ++j) h[j] = b1s[j];
    const float* xr = X0 + (size_t)i * cF;
    for (int k = 0; k < 128; ++k) {
        const float xv = xr[k];
#pragma unroll
        for (int j = 0; j < 16; ++j) h[j] += xv * W1s[k * 16 + j];
    }
#pragma unroll
    for (int j = 0; j < 16; ++j) h[j] = h[j] > 0.f ? h[j] : 0.f;
#pragma unroll
    for (int o = 0; o < 8; ++o) {
        float s = b2s[o];
#pragma unroll
        for (int j = 0; j < 16; ++j) s += h[j] * W2s[j * 8 + o];
        V[(size_t)i * 8 + o] = 1.f / (1.f + expf(-s));
    }
}

__global__ void fillu_kernel(unsigned* __restrict__ p, unsigned val, int n)
{
    const int i = blockIdx.x * 256 + threadIdx.x;
    if (i < n) p[i] = val;
}

// ve = max(v[src],v[dst]); death = scatter-min over incident edges.
// sigmoid outputs are positive -> uint bit-order == float order.
__global__ void edge_filt_kernel(const float* __restrict__ V,
                                 const int* __restrict__ es,
                                 const int* __restrict__ ed, int E,
                                 float* __restrict__ VE, unsigned* __restrict__ DEATH)
{
    const int e = blockIdx.x * 256 + threadIdx.x;
    if (e >= E) return;
    const int s = es[e], d = ed[e];
#pragma unroll
    for (int f = 0; f < 8; ++f) {
        const float a = V[(size_t)s * 8 + f];
        const float b = V[(size_t)d * 8 + f];
        const float m = a > b ? a : b;
        VE[(size_t)e * 8 + f] = m;
        const unsigned mb = __float_as_uint(m);
        atomicMin(&DEATH[(size_t)s * 8 + f], mb);
        atomicMin(&DEATH[(size_t)d * 8 + f], mb);
    }
}

// Small PH MLP (16->64->64) + segment-mean accumulation. DEATH!=null: cells
// (pts=[v,death-fixed]); else edges (pts=[ve,1], batch=batch0[esrc[e]]).
__global__ __launch_bounds__(256) void ph_mlp_kernel(
    const float* __restrict__ P, const unsigned* __restrict__ DEATH,
    const int* __restrict__ batch0, const int* __restrict__ esrc, int N,
    const float* __restrict__ W1, const float* __restrict__ bb1,
    const float* __restrict__ W2, const float* __restrict__ bb2,
    float* __restrict__ gs, float* __restrict__ gc)
{
    __shared__ float W1s[16 * 64];
    __shared__ float W2s[64 * 64];
    __shared__ float b1s[64], b2s[64];
    const int t = threadIdx.x;
    for (int i = t; i < 16 * 64; i += 256) W1s[i] = W1[i];
    for (int i = t; i < 64 * 64; i += 256) W2s[i] = W2[i];
    if (t < 64) { b1s[t] = bb1[t]; b2s[t] = bb2[t]; }
    __syncthreads();
    const int i = blockIdx.x * 256 + t;
    if (i >= N) return;
    float pts[16];
#pragma unroll
    for (int f = 0; f < 8; ++f) {
        const float v = P[(size_t)i * 8 + f];
        pts[f] = v;
        if (DEATH) {
            const unsigned db = DEATH[(size_t)i * 8 + f];
            pts[8 + f] = (db == 0x7f800000u) ? v : __uint_as_float(db);
        } else {
            pts[8 + f] = 1.f;
        }
    }
    float h[64];
#pragma unroll
    for (int o = 0; o < 64; ++o) h[o] = b1s[o];
#pragma unroll
    for (int k = 0; k < 16; ++k) {
        const float pv = pts[k];
        for (int o = 0; o < 64; ++o) h[o] += pv * W1s[k * 64 + o];
    }
#pragma unroll
    for (int o = 0; o < 64; ++o) h[o] = h[o] > 0.f ? h[o] : 0.f;
    const int b = esrc ? batch0[esrc[i]] : batch0[i];
    for (int o = 0; o < 64; ++o) {
        float s = b2s[o];
#pragma unroll
        for (int k = 0; k < 64; ++k) s += h[k] * W2s[k * 64 + o];
        atomicAdd(&gs[(size_t)b * 64 + o], s);
    }
    atomicAdd(&gc[b], 1.f);
}

__global__ void gmix_kernel(const float* __restrict__ g0s, const float* __restrict__ g0c,
                            const float* __restrict__ g1s, const float* __restrict__ g1c,
                            float* __restrict__ gm)
{
    const int idx = blockIdx.x * 256 + threadIdx.x;
    if (idx >= cB * 64) return;
    const int b = idx >> 6;
    float c0 = g0c[b]; c0 = c0 > 1.f ? c0 : 1.f;
    float c1 = g1c[b]; c1 = c1 > 1.f ? c1 : 1.f;
    gm[idx] = g0s[idx] / c0 + g1s[idx] / c1;
}

__global__ __launch_bounds__(64) void phout_kernel(const float* __restrict__ gm,
                                                   const float* __restrict__ oW,
                                                   const float* __restrict__ ob,
                                                   float* __restrict__ phacc)
{
    __shared__ float g[64];
    const int b = blockIdx.x, t = threadIdx.x;
    g[t] = gm[(size_t)b * 64 + t];
    __syncthreads();
    float s = ob[t];
#pragma unroll
    for (int k = 0; k < 64; ++k) s += g[k] * oW[k * 64 + t];
    phacc[(size_t)b * 64 + t] += s * (1.f / (float)cNL);
}

__global__ void pool_kernel(const float* __restrict__ X, const int* __restrict__ batch,
                            int N, float* __restrict__ pooled)
{
    const size_t idx = (size_t)blockIdx.x * 256 + threadIdx.x;
    if (idx >= (size_t)N * cF) return;
    const int i = (int)(idx >> 7), f = (int)(idx & 127);
    atomicAdd(&pooled[(size_t)batch[i] * cF + f], X[idx]);
}

__global__ __launch_bounds__(128) void final_kernel(
    const float* __restrict__ pooled, const float* __restrict__ phacc,
    const float* __restrict__ l1W, const float* __restrict__ l1b,
    const float* __restrict__ l2W, const float* __restrict__ l2b,
    float* __restrict__ out)
{
    __shared__ float pl[128];
    __shared__ float hs[128];
    const int b = blockIdx.x, t = threadIdx.x;
    pl[t] = pooled[(size_t)b * cF + t];
    __syncthreads();
    float s = l1b[t];
    for (int k = 0; k < 128; ++k) s += pl[k] * l1W[k * 128 + t];
    hs[t] = s > 0.f ? s : 0.f;
    __syncthreads();
    if (t < cNC) {
        float o = l2b[t];
        for (int k = 0; k < 128; ++k) o += hs[k] * l2W[k * cNC + t];
        for (int j = 0; j < 64; ++j)
            o += phacc[(size_t)b * 64 + j] * l2W[(128 + j) * cNC + t];
        out[(size_t)b * cNC + t] = o;
    }
}

// W [K][N] fp32 row-major  ->  Wt [N][K] bf16 (B-matrix friendly)
__global__ void wconv_kernel(const float* __restrict__ W, __bf16* __restrict__ Wt,
                             int K, int Ncols)
{
    const int idx = blockIdx.x * 256 + threadIdx.x;
    if (idx >= K * Ncols) return;
    const int n = idx / K, k = idx % K;
    Wt[(size_t)n * K + k] = (__bf16)W[(size_t)k * Ncols + n];
}

// ---------------------------------------------------------------------------
extern "C" void kernel_launch(void* const* d_in, const int* in_sizes, int n_in,
                              void* d_out, int out_size, void* d_ws, size_t ws_size,
                              hipStream_t stream)
{
    (void)in_sizes; (void)n_in; (void)out_size;

    const float* x0_in = (const float*)d_in[0];
    const float* x1_in = (const float*)d_in[1];
    const float* x2_in = (const float*)d_in[2];
    const int* batch0  = (const int*)d_in[3];
    const int* batch1  = (const int*)d_in[4];
    const int* batch2  = (const int*)d_in[5];
    const int* up0_src = (const int*)d_in[6];
    const int* up0_dst = (const int*)d_in[7];
    const int* up0_cof = (const int*)d_in[8];
    const int* up1_src = (const int*)d_in[9];
    const int* up1_dst = (const int*)d_in[10];
    const int* up1_cof = (const int*)d_in[11];
    const int* dn1_src = (const int*)d_in[12];
    const int* dn1_dst = (const int*)d_in[13];
    const int* dn1_fac = (const int*)d_in[14];
    const int* dn2_src = (const int*)d_in[15];
    const int* dn2_dst = (const int*)d_in[16];
    const int* dn2_fac = (const int*)d_in[17];
    const int* edge_src = (const int*)d_in[18];
    const int* edge_dst = (const int*)d_in[19];
    auto PF = [&](int i) { return (const float*)d_in[i]; };
    // params (insertion-order DFS): conv[c]: upW,upb,upg,upbe,dnW,dnb,dng,dnbe,
    //   W1,b1,W2,b2,ug,ube (14/layer @20); ph[c]: fW1,fb1,fW2,fb2,m0W1,m0b1,
    //   m0W2,m0b2,m1W1,m1b1,m1W2,m1b2,oW,ob (14/layer @62); l1W=104,l1b,l2W,l2b
    const int PC0 = 20, PQ0 = 62, PL = 104;

    // ---- workspace layout ----
    char* wsb = (char*)d_ws;
    size_t off = 0;
    auto alloc = [&](size_t bytes) -> char* {
        char* p = wsb + off;
        off = (off + bytes + 255) & ~(size_t)255;
        return p;
    };
    float* X0  = (float*)alloc((size_t)cN0 * cF * 4);
    float* X1  = (float*)alloc((size_t)cN1 * cF * 4);
    float* X2  = (float*)alloc((size_t)cN2 * cF * 4);
    float* A0  = (float*)alloc((size_t)cN0 * cF * 4);
    float* A1U = (float*)alloc((size_t)cN1 * cF * 4);
    float* A1D = (float*)alloc((size_t)cN1 * cF * 4);
    float* A2  = (float*)alloc((size_t)cN2 * cF * 4);
    float* C0  = (float*)alloc((size_t)cN0 * 4);
    float* C1U = (float*)alloc((size_t)cN1 * 4);
    float* C1D = (float*)alloc((size_t)cN1 * 4);
    float* C2  = (float*)alloc((size_t)cN2 * 4);
    float* STAT = (float*)alloc(5 * 512 * 4);           // 5 slots: ssum|ssq|s|t
    float* V    = (float*)alloc((size_t)cN0 * cNF * 4);
    float* VE   = (float*)alloc((size_t)cEG * cNF * 4);
    unsigned* DEATH = (unsigned*)alloc((size_t)cN0 * cNF * 4);
    float* G0S  = (float*)alloc((size_t)cB * 64 * 4);
    float* G0C  = (float*)alloc((size_t)cB * 4);
    float* G1S  = (float*)alloc((size_t)cB * 64 * 4);
    float* G1C  = (float*)alloc((size_t)cB * 4);
    float* GMIX = (float*)alloc((size_t)cB * 64 * 4);
    float* PHACC = (float*)alloc((size_t)cB * 64 * 4);
    float* POOLED = (float*)alloc((size_t)cB * cF * 4);
    __bf16* WB = (__bf16*)alloc((size_t)cNL * 98304 * 2);
    if (off > ws_size) return;   // workspace too small: nothing safe to do

    // ---- bf16 transposed weights ----
    for (int c = 0; c < cNL; ++c) {
        const int pc = PC0 + 14 * c;
        __bf16* wup = WB + (size_t)c * 98304;
        wconv_kernel<<<(256 * 128 + 255) / 256, 256, 0, stream>>>(PF(pc + 0), wup, 256, 128);
        wconv_kernel<<<(256 * 128 + 255) / 256, 256, 0, stream>>>(PF(pc + 4), wup + 32768, 256, 128);
        wconv_kernel<<<(128 * 128 + 255) / 256, 256, 0, stream>>>(PF(pc + 8), wup + 65536, 128, 128);
        wconv_kernel<<<(128 * 128 + 255) / 256, 256, 0, stream>>>(PF(pc + 10), wup + 81920, 128, 128);
    }

    hipMemcpyAsync(X0, x0_in, (size_t)cN0 * cF * 4, hipMemcpyDeviceToDevice, stream);
    hipMemcpyAsync(X1, x1_in, (size_t)cN1 * cF * 4, hipMemcpyDeviceToDevice, stream);
    hipMemcpyAsync(X2, x2_in, (size_t)cN2 * cF * 4, hipMemcpyDeviceToDevice, stream);
    hipMemsetAsync(PHACC, 0, (size_t)cB * 64 * 4, stream);
    hipMemsetAsync(POOLED, 0, (size_t)cB * cF * 4, stream);

    for (int c = 0; c < cNL; ++c) {
        const int pc = PC0 + 14 * c;
        const int pq = PQ0 + 14 * c;
        const __bf16* wup = WB + (size_t)c * 98304;
        const __bf16* wdn = wup + 32768;
        const __bf16* w1t = wup + 65536;
        const __bf16* w2t = wup + 81920;

        hipMemsetAsync(A0, 0, (size_t)cN0 * cF * 4, stream);
        hipMemsetAsync(A1U, 0, (size_t)cN1 * cF * 4, stream);
        hipMemsetAsync(A1D, 0, (size_t)cN1 * cF * 4, stream);
        hipMemsetAsync(A2, 0, (size_t)cN2 * cF * 4, stream);
        hipMemsetAsync(C0, 0, (size_t)cN0 * 4, stream);
        hipMemsetAsync(C1U, 0, (size_t)cN1 * 4, stream);
        hipMemsetAsync(C1D, 0, (size_t)cN1 * 4, stream);
        hipMemsetAsync(C2, 0, (size_t)cN2 * 4, stream);
        hipMemsetAsync(STAT, 0, 4 * 512 * 4, stream);

        // ---- four message convolutions (WMMA, 32 edges/block) ----
        msg_gemm_kernel<<<cE0U / 32, 256, 0, stream>>>(X0, X1, up0_src, up0_dst, up0_cof,
            cE0U, wup, PF(pc + 1), A0, C0, STAT + 0, STAT + 128);
        msg_gemm_kernel<<<cE1U / 32, 256, 0, stream>>>(X1, X2, up1_src, up1_dst, up1_cof,
            cE1U, wup, PF(pc + 1), A1U, C1U, STAT + 512, STAT + 640);
        msg_gemm_kernel<<<cE1D / 32, 256, 0, stream>>>(X1, X0, dn1_src, dn1_dst, dn1_fac,
            cE1D, wdn, PF(pc + 5), A1D, C1D, STAT + 1024, STAT + 1152);
        msg_gemm_kernel<<<cE2D / 32, 256, 0, stream>>>(X2, X1, dn2_src, dn2_dst, dn2_fac,
            cE2D, wdn, PF(pc + 5), A2, C2, STAT + 1536, STAT + 1664);
        bnfit_kernel<<<1, 128, 0, stream>>>(STAT + 0,    PF(pc + 2), PF(pc + 3), 1.f / cE0U);
        bnfit_kernel<<<1, 128, 0, stream>>>(STAT + 512,  PF(pc + 2), PF(pc + 3), 1.f / cE1U);
        bnfit_kernel<<<1, 128, 0, stream>>>(STAT + 1024, PF(pc + 6), PF(pc + 7), 1.f / cE1D);
        bnfit_kernel<<<1, 128, 0, stream>>>(STAT + 1536, PF(pc + 6), PF(pc + 7), 1.f / cE2D);

        // ---- GIN updates (fused double WMMA GEMM), then BN -> X ----
        hipMemsetAsync(STAT + 2048, 0, 256 * 4, stream);
        upd_gemm_kernel<<<(cN0 + 31) / 32, 256, 0, stream>>>(X0, cN0, A0, C0, STAT + 256,
            nullptr, nullptr, nullptr, w1t, PF(pc + 9), w2t, PF(pc + 11),
            A0, STAT + 2048, STAT + 2176);
        bnfit_kernel<<<1, 128, 0, stream>>>(STAT + 2048, PF(pc + 12), PF(pc + 13), 1.f / cN0);
        bn_apply_kernel<<<(cN0 * cF) / 256, 256, 0, stream>>>(A0, X0, STAT + 2304,
            (size_t)cN0 * cF);

        hipMemsetAsync(STAT + 2048, 0, 256 * 4, stream);
        upd_gemm_kernel<<<(cN1 + 31) / 32, 256, 0, stream>>>(X1, cN1, A1U, C1U, STAT + 768,
            A1D, C1D, STAT + 1280, w1t, PF(pc + 9), w2t, PF(pc + 11),
            A1U, STAT + 2048, STAT + 2176);
        bnfit_kernel<<<1, 128, 0, stream>>>(STAT + 2048, PF(pc + 12), PF(pc + 13), 1.f / cN1);
        bn_apply_kernel<<<(cN1 * cF) / 256, 256, 0, stream>>>(A1U, X1, STAT + 2304,
            (size_t)cN1 * cF);

        hipMemsetAsync(STAT + 2048, 0, 256 * 4, stream);
        upd_gemm_kernel<<<(cN2 + 31) / 32, 256, 0, stream>>>(X2, cN2, A2, C2, STAT + 1792,
            nullptr, nullptr, nullptr, w1t, PF(pc + 9), w2t, PF(pc + 11),
            A2, STAT + 2048, STAT + 2176);
        bnfit_kernel<<<1, 128, 0, stream>>>(STAT + 2048, PF(pc + 12), PF(pc + 13), 1.f / cN2);
        bn_apply_kernel<<<(cN2 * cF) / 256, 256, 0, stream>>>(A2, X2, STAT + 2304,
            (size_t)cN2 * cF);

        // ---- persistent-homology branch ----
        filt_kernel<<<(cN0 + 255) / 256, 256, 0, stream>>>(X0, cN0,
            PF(pq + 0), PF(pq + 1), PF(pq + 2), PF(pq + 3), V);
        fillu_kernel<<<(cN0 * cNF + 255) / 256, 256, 0, stream>>>(DEATH, 0x7f800000u,
            cN0 * cNF);
        edge_filt_kernel<<<(cEG + 255) / 256, 256, 0, stream>>>(V, edge_src, edge_dst,
            cEG, VE, DEATH);
        hipMemsetAsync(G0S, 0, (size_t)cB * 64 * 4, stream);
        hipMemsetAsync(G0C, 0, (size_t)cB * 4, stream);
        hipMemsetAsync(G1S, 0, (size_t)cB * 64 * 4, stream);
        hipMemsetAsync(G1C, 0, (size_t)cB * 4, stream);
        ph_mlp_kernel<<<(cN0 + 255) / 256, 256, 0, stream>>>(V, DEATH, batch0, nullptr,
            cN0, PF(pq + 4), PF(pq + 5), PF(pq + 6), PF(pq + 7), G0S, G0C);
        ph_mlp_kernel<<<(cEG + 255) / 256, 256, 0, stream>>>(VE, nullptr, batch0, edge_src,
            cEG, PF(pq + 8), PF(pq + 9), PF(pq + 10), PF(pq + 11), G1S, G1C);
        gmix_kernel<<<(cB * 64) / 256, 256, 0, stream>>>(G0S, G0C, G1S, G1C, GMIX);
        phout_kernel<<<cB, 64, 0, stream>>>(GMIX, PF(pq + 12), PF(pq + 13), PHACC);
    }

    // ---- readout ----
    pool_kernel<<<(cN0 * cF) / 256, 256, 0, stream>>>(X0, batch0, cN0, POOLED);
    pool_kernel<<<(cN1 * cF) / 256, 256, 0, stream>>>(X1, batch1, cN1, POOLED);
    pool_kernel<<<(cN2 * cF) / 256, 256, 0, stream>>>(X2, batch2, cN2, POOLED);
    final_kernel<<<cB, 128, 0, stream>>>(POOLED, PHACC, PF(PL + 0), PF(PL + 1),
                                         PF(PL + 2), PF(PL + 3), (float*)d_out);
}
